// AttnDecoderLSTM3L_R4_Rescon_WinATT_12369505813158
// MI455X (gfx1250) — compile-verified, matrix operand order checked by reading, not run on hardware
//
#include <hip/hip_runtime.h>
#include <math.h>

#define Hdim 2048
#define INdim 1024
#define OUTdim 320
#define Tdim 4096

typedef __attribute__((ext_vector_type(16))) __bf16 v16bf;
typedef __attribute__((ext_vector_type(8)))  float  v8f;

// Branch-free fast transcendentals (v_exp_f32 + v_rcp_f32, no EXEC divergence)
__device__ __forceinline__ float sigmoid_fast(float v)
{
    return 1.f / (1.f + __expf(-v));
}
__device__ __forceinline__ float tanh_fast(float v)
{
    const float x = fminf(fmaxf(v, -15.f), 15.f);   // e^{30} finite, no inf/nan
    const float e = __expf(2.f * x);
    return (e - 1.f) / (e + 1.f);
}

// ---------------------------------------------------------------------------
// Generic memory-bound GEMV: one wave per row, b128 loads, wave shuffle reduce
// y[row] = act( dot(W[row,:], x) + b[row] )
// ---------------------------------------------------------------------------
__global__ void __launch_bounds__(256)
gemv_kernel(const float* __restrict__ W, const float* __restrict__ x,
            const float* __restrict__ b, float* __restrict__ y,
            int rows, int K, int relu)
{
    const int lane = threadIdx.x & 31;
    const int wave = threadIdx.x >> 5;
    const int row  = blockIdx.x * 8 + wave;
    if (row >= rows) return;
    const float* w = W + (size_t)row * K;
    float sum = 0.f;
#pragma unroll 2
    for (int i = lane * 4; i < K; i += 128) {
        float4 wv = *(const float4*)(w + i);
        float4 xv = *(const float4*)(x + i);
        sum += wv.x * xv.x + wv.y * xv.y + wv.z * xv.z + wv.w * xv.w;
    }
    for (int off = 1; off < 32; off <<= 1) sum += __shfl_xor(sum, off, 32);
    if (lane == 0) {
        if (b)    sum += b[row];
        if (relu) sum = fmaxf(sum, 0.f);
        y[row] = sum;
    }
}

// ---------------------------------------------------------------------------
// LSTM gate GEMV: g[row] = dot(Wx[row],x) + dot(Wh[row],h) + b1[row] + b2[row]
// ---------------------------------------------------------------------------
__global__ void __launch_bounds__(256)
gemv2_kernel(const float* __restrict__ Wx, const float* __restrict__ x, int Kx,
             const float* __restrict__ Wh, const float* __restrict__ h, int Kh,
             const float* __restrict__ b1, const float* __restrict__ b2,
             float* __restrict__ g, int rows)
{
    const int lane = threadIdx.x & 31;
    const int wave = threadIdx.x >> 5;
    const int row  = blockIdx.x * 8 + wave;
    if (row >= rows) return;
    const float* wx = Wx + (size_t)row * Kx;
    const float* wh = Wh + (size_t)row * Kh;
    float sum = 0.f;
#pragma unroll 2
    for (int i = lane * 4; i < Kx; i += 128) {
        float4 wv = *(const float4*)(wx + i);
        float4 xv = *(const float4*)(x + i);
        sum += wv.x * xv.x + wv.y * xv.y + wv.z * xv.z + wv.w * xv.w;
    }
#pragma unroll 2
    for (int i = lane * 4; i < Kh; i += 128) {
        float4 wv = *(const float4*)(wh + i);
        float4 xv = *(const float4*)(h + i);
        sum += wv.x * xv.x + wv.y * xv.y + wv.z * xv.z + wv.w * xv.w;
    }
    for (int off = 1; off < 32; off <<= 1) sum += __shfl_xor(sum, off, 32);
    if (lane == 0) g[row] = sum + b1[row] + b2[row];
}

// ---------------------------------------------------------------------------
// LSTM cell pointwise update (PyTorch gate order: i, f, g, o)
// ---------------------------------------------------------------------------
__global__ void __launch_bounds__(256)
lstm_update_kernel(const float* __restrict__ g, const float* __restrict__ c_in,
                   float* __restrict__ h_out, float* __restrict__ c_out)
{
    const int j = blockIdx.x * 256 + threadIdx.x;
    if (j >= Hdim) return;
    const float gi = g[j];
    const float gf = g[Hdim + j];
    const float gg = g[2 * Hdim + j];
    const float go = g[3 * Hdim + j];
    const float c2 = sigmoid_fast(gf) * c_in[j] + sigmoid_fast(gi) * tanh_fast(gg);
    c_out[j] = c2;
    h_out[j] = sigmoid_fast(go) * tanh_fast(c2);
}

// ---------------------------------------------------------------------------
// Fused attention-score GEMM on the matrix pipe:
//   gamma[t] = sum_h tanh( enc[t,:]·attn2_w[h,:] + attn2_b[h] + s[h] ) * w3[h]
// WMMA bf16 tiles, fp32->bf16 conversion in registers; attn2_w re-reads hit L2.
// k-loop unrolled 2x so two independent load->cvt->wmma chains pipeline.
// Deterministic: each wave writes private partials to gpart[block*128+wave*16+m].
// ---------------------------------------------------------------------------
__device__ __forceinline__ void pack8(v16bf& dst, int base, float4 a, float4 b)
{
    dst[base + 0] = (__bf16)a.x; dst[base + 1] = (__bf16)a.y;
    dst[base + 2] = (__bf16)a.z; dst[base + 3] = (__bf16)a.w;
    dst[base + 4] = (__bf16)b.x; dst[base + 5] = (__bf16)b.y;
    dst[base + 6] = (__bf16)b.z; dst[base + 7] = (__bf16)b.w;
}

__device__ __forceinline__ v8f wmma_step(const float* __restrict__ arow,
                                         const float* __restrict__ brow,
                                         int kb, v8f acc)
{
    // A fragment: 16-bit A 16x32 layout (elems 0..7 = K kb..kb+7,
    // elems 8..15 = K kb+16..kb+23 for this lane-half)
    float4 a0 = *(const float4*)(arow + kb);
    float4 a1 = *(const float4*)(arow + kb + 4);
    float4 a2 = *(const float4*)(arow + kb + 16);
    float4 a3 = *(const float4*)(arow + kb + 20);
    float4 b0 = *(const float4*)(brow + kb);
    float4 b1 = *(const float4*)(brow + kb + 4);
    float4 b2 = *(const float4*)(brow + kb + 16);
    float4 b3 = *(const float4*)(brow + kb + 20);

    v16bf A, B;
    pack8(A, 0, a0, a1);
    pack8(A, 8, a2, a3);
    pack8(B, 0, b0, b1);
    pack8(B, 8, b2, b3);

    return __builtin_amdgcn_wmma_f32_16x16x32_bf16(
        false, A, false, B, (short)0, acc, false, false);
}

__global__ void __launch_bounds__(256)
attn_gamma_wmma(const float* __restrict__ enc, const float* __restrict__ w2,
                const float* __restrict__ b2, const float* __restrict__ s,
                const float* __restrict__ w3, float* __restrict__ gpart)
{
    const int lane = threadIdx.x & 31;
    const int wave = threadIdx.x >> 5;
    const int n    = lane & 15;   // tile column (N) / fragment row selector
    const int hf   = lane >> 4;   // lane half -> K sub-range selector
    const int t0   = blockIdx.x * 16;

    const float* arow = enc + (size_t)(t0 + n) * Hdim;

    float gp[8];
#pragma unroll
    for (int v = 0; v < 8; ++v) gp[v] = 0.f;

    for (int ht = wave; ht < Hdim / 16; ht += 8) {
        const int h0 = ht * 16;
        const float* brow = w2 + (size_t)(h0 + n) * Hdim;
        v8f acc0 = (v8f){0.f, 0.f, 0.f, 0.f, 0.f, 0.f, 0.f, 0.f};
        v8f acc1 = (v8f){0.f, 0.f, 0.f, 0.f, 0.f, 0.f, 0.f, 0.f};

        // two independent accumulator chains -> pipelined loads/cvt/wmma
        for (int k = 0; k < Hdim; k += 64) {
            const int kb = k + hf * 8;
            acc0 = wmma_step(arow, brow, kb,      acc0);
            acc1 = wmma_step(arow, brow, kb + 32, acc1);
        }

        const int   h    = h0 + n;
        const float bias = b2[h] + s[h];
        const float w3h  = w3[h];
#pragma unroll
        for (int v = 0; v < 8; ++v)
            gp[v] += tanh_fast(acc0[v] + acc1[v] + bias) * w3h;
    }

    // reduce the h (column) dimension across the 16 lanes of each half
#pragma unroll
    for (int v = 0; v < 8; ++v) {
        float t = gp[v];
        t += __shfl_xor(t, 1, 32);
        t += __shfl_xor(t, 2, 32);
        t += __shfl_xor(t, 4, 32);
        t += __shfl_xor(t, 8, 32);
        gp[v] = t;
    }
    if (n == 0) {
#pragma unroll
        for (int v = 0; v < 8; ++v) {
            const int m = v + 8 * hf;               // D layout: M = v + 8*half
            gpart[(size_t)blockIdx.x * 128 + wave * 16 + m] = gp[v];
        }
    }
}

// ---------------------------------------------------------------------------
// Reduce per-wave gamma partials (fixed order) + softmax over T, single block
// ---------------------------------------------------------------------------
__global__ void __launch_bounds__(1024)
softmax_kernel(const float* __restrict__ gpart, float* __restrict__ aw)
{
    __shared__ float g[Tdim];
    __shared__ float red[32];
    const int tid = threadIdx.x;

    for (int t = tid; t < Tdim; t += 1024) {
        const int blk = t >> 4, m = t & 15;
        float ssum = 0.f;
#pragma unroll
        for (int w = 0; w < 8; ++w) ssum += gpart[blk * 128 + w * 16 + m];
        g[t] = ssum;
    }
    __syncthreads();

    // block max
    float mx = -INFINITY;
    for (int t = tid; t < Tdim; t += 1024) mx = fmaxf(mx, g[t]);
    for (int off = 1; off < 32; off <<= 1) mx = fmaxf(mx, __shfl_xor(mx, off, 32));
    if ((tid & 31) == 0) red[tid >> 5] = mx;
    __syncthreads();
    if (tid < 32) {
        float v = red[tid];
        for (int off = 1; off < 32; off <<= 1) v = fmaxf(v, __shfl_xor(v, off, 32));
        if (tid == 0) red[0] = v;
    }
    __syncthreads();
    mx = red[0];
    __syncthreads();

    // exp + block sum
    float lsum = 0.f;
    for (int t = tid; t < Tdim; t += 1024) {
        const float e = __expf(g[t] - mx);
        g[t] = e;
        lsum += e;
    }
    for (int off = 1; off < 32; off <<= 1) lsum += __shfl_xor(lsum, off, 32);
    if ((tid & 31) == 0) red[tid >> 5] = lsum;
    __syncthreads();
    if (tid < 32) {
        float v = red[tid];
        for (int off = 1; off < 32; off <<= 1) v += __shfl_xor(v, off, 32);
        if (tid == 0) red[0] = v;
    }
    __syncthreads();
    const float inv = 1.f / red[0];
    for (int t = tid; t < Tdim; t += 1024) aw[t] = g[t] * inv;
}

// ---------------------------------------------------------------------------
// attn_applied[h] = sum_t aw[t] * enc[t,h]  (coalesced across threads),
// and build o3 = [o2, attn_applied]
// ---------------------------------------------------------------------------
__global__ void __launch_bounds__(256)
attn_apply_kernel(const float* __restrict__ aw, const float* __restrict__ enc,
                  const float* __restrict__ o2, float* __restrict__ aa,
                  float* __restrict__ o3)
{
    const int h = blockIdx.x * 256 + threadIdx.x;
    if (h >= Hdim) return;
    float sum = 0.f;
    const float* col = enc + h;
    for (int t = 0; t < Tdim; t += 4) {
        sum += aw[t + 0] * col[(size_t)(t + 0) * Hdim];
        sum += aw[t + 1] * col[(size_t)(t + 1) * Hdim];
        sum += aw[t + 2] * col[(size_t)(t + 2) * Hdim];
        sum += aw[t + 3] * col[(size_t)(t + 3) * Hdim];
    }
    aa[h]        = sum;
    o3[h]        = o2[h];
    o3[Hdim + h] = sum;
}

// o6 = concat(o5, attn_applied) + o3
__global__ void __launch_bounds__(256)
res_concat_kernel(const float* __restrict__ o5, const float* __restrict__ aa,
                  const float* __restrict__ o3, float* __restrict__ o6)
{
    const int j = blockIdx.x * 256 + threadIdx.x;
    if (j >= 2 * Hdim) return;
    const float a = (j < Hdim) ? o5[j] : aa[j - Hdim];
    o6[j] = a + o3[j];
}

// ---------------------------------------------------------------------------
extern "C" void kernel_launch(void* const* d_in, const int* in_sizes, int n_in,
                              void* d_out, int out_size, void* d_ws, size_t ws_size,
                              hipStream_t stream)
{
    (void)in_sizes; (void)n_in; (void)out_size; (void)ws_size;

    const float* x       = (const float*)d_in[0];
    const float* h1      = (const float*)d_in[1];
    const float* c1      = (const float*)d_in[2];
    const float* h2      = (const float*)d_in[3];
    const float* c2i     = (const float*)d_in[4];
    const float* h3      = (const float*)d_in[5];
    const float* c3i     = (const float*)d_in[6];
    const float* enc     = (const float*)d_in[7];
    // d_in[8..10]: win_len, dec_ix, sl_dec (unused at inference)
    const float* pn1_w   = (const float*)d_in[11];
    const float* pn1_b   = (const float*)d_in[12];
    const float* pn2_w   = (const float*)d_in[13];
    const float* pn2_b   = (const float*)d_in[14];
    const float* attn1_w = (const float*)d_in[15];
    const float* attn2_w = (const float*)d_in[16];
    const float* attn2_b = (const float*)d_in[17];
    const float* attn3_w = (const float*)d_in[18];
    const float* l1_wih  = (const float*)d_in[19];
    const float* l1_whh  = (const float*)d_in[20];
    const float* l1_bih  = (const float*)d_in[21];
    const float* l1_bhh  = (const float*)d_in[22];
    const float* l2_wih  = (const float*)d_in[23];
    const float* l2_whh  = (const float*)d_in[24];
    const float* l2_bih  = (const float*)d_in[25];
    const float* l2_bhh  = (const float*)d_in[26];
    const float* l3_wih  = (const float*)d_in[27];
    const float* l3_whh  = (const float*)d_in[28];
    const float* l3_bih  = (const float*)d_in[29];
    const float* l3_bhh  = (const float*)d_in[30];
    const float* lin1_w  = (const float*)d_in[31];
    const float* lin1_b  = (const float*)d_in[32];
    const float* lin2_w  = (const float*)d_in[33];
    const float* lin2_b  = (const float*)d_in[34];
    const float* lin3_w  = (const float*)d_in[35];
    const float* lin3_b  = (const float*)d_in[36];
    const float* lin4_w  = (const float*)d_in[37];
    const float* lin4_b  = (const float*)d_in[38];

    float* out = (float*)d_out;
    // d_out layout (flat, return order):
    float* o13 = out + 1280;
    float* o2  = out + 1600;
    float* c1n = out + 3648;
    float* o5  = out + 5696;
    float* c2n = out + 7744;
    float* o9  = out + 9792;
    float* c3n = out + 11840;
    float* aw  = out + 13888;

    float* ws    = (float*)d_ws;
    float* y1    = ws;           // 2048
    float* p     = ws + 2048;    // 2048
    float* gates = ws + 4096;    // 8192
    float* s     = ws + 12288;   // 2048
    float* aa    = ws + 14336;   // 2048
    float* o3    = ws + 16384;   // 4096
    float* o6    = ws + 20480;   // 4096
    float* gpart = ws + 24576;   // 256 * 128 = 32768

    // prenet
    gemv_kernel<<<256, 256, 0, stream>>>(pn1_w, x,  pn1_b, y1, Hdim, INdim, 1);
    gemv_kernel<<<256, 256, 0, stream>>>(pn2_w, y1, pn2_b, p,  Hdim, Hdim,  1);

    // LSTM layer 1
    gemv2_kernel<<<1024, 256, 0, stream>>>(l1_wih, p, Hdim, l1_whh, h1, Hdim,
                                           l1_bih, l1_bhh, gates, 4 * Hdim);
    lstm_update_kernel<<<8, 256, 0, stream>>>(gates, c1, o2, c1n);

    // attention: s = attn1_w @ o2, then fused WMMA score GEMM + softmax + apply
    gemv_kernel<<<256, 256, 0, stream>>>(attn1_w, o2, nullptr, s, Hdim, Hdim, 0);
    attn_gamma_wmma<<<Tdim / 16, 256, 0, stream>>>(enc, attn2_w, attn2_b, s,
                                                   attn3_w, gpart);
    softmax_kernel<<<1, 1024, 0, stream>>>(gpart, aw);
    attn_apply_kernel<<<8, 256, 0, stream>>>(aw, enc, o2, aa, o3);

    // LSTM layer 2
    gemv2_kernel<<<1024, 256, 0, stream>>>(l2_wih, o3, 2 * Hdim, l2_whh, h2, Hdim,
                                           l2_bih, l2_bhh, gates, 4 * Hdim);
    lstm_update_kernel<<<8, 256, 0, stream>>>(gates, c2i, o5, c2n);

    // residual concat
    res_concat_kernel<<<16, 256, 0, stream>>>(o5, aa, o3, o6);

    // LSTM layer 3
    gemv2_kernel<<<1024, 256, 0, stream>>>(l3_wih, o6, 2 * Hdim, l3_whh, h3, Hdim,
                                           l3_bih, l3_bhh, gates, 4 * Hdim);
    lstm_update_kernel<<<8, 256, 0, stream>>>(gates, c3i, o9, c3n);

    // output heads: output = concat(lin1..lin4), plus o13 = lin4 result
    gemv_kernel<<<40, 256, 0, stream>>>(lin1_w, o9, lin1_b, out + 0,   OUTdim, Hdim, 0);
    gemv_kernel<<<40, 256, 0, stream>>>(lin2_w, o9, lin2_b, out + 320, OUTdim, Hdim, 0);
    gemv_kernel<<<40, 256, 0, stream>>>(lin3_w, o9, lin3_b, out + 640, OUTdim, Hdim, 0);
    gemv_kernel<<<40, 256, 0, stream>>>(lin4_w, o9, lin4_b, out + 960, OUTdim, Hdim, 0);
    gemv_kernel<<<40, 256, 0, stream>>>(lin4_w, o9, lin4_b, o13,       OUTdim, Hdim, 0);
}
